// MixLinear_GEMM_27814208209058
// MI455X (gfx1250) — 8-bit matrix operand order repaired and checked by reading, compiled, not run on hardware
//
#include <hip/hip_runtime.h>
#include <hip/hip_fp16.h>

typedef __attribute__((ext_vector_type(8))) int v8i;
typedef __attribute__((ext_vector_type(4))) int v4i;

#define SIGMA 20.0f
#define MAXOUT 32

__global__ void k_init(int* mask, int K) {
    int k = blockIdx.x * 256 + threadIdx.x;
    if (k < K) mask[k] = 0;
}

__global__ void k_outlier(const __half* __restrict__ x, int* __restrict__ mask,
                          int M, int K, int rows_per_block) {
    int k = blockIdx.x * 256 + threadIdx.x;
    if (k >= K) return;
    int m0 = blockIdx.y * rows_per_block;
    int m1 = m0 + rows_per_block; if (m1 > M) m1 = M;
    bool f = false;
    for (int m = m0; m < m1; ++m) {
        float v = __half2float(x[(size_t)m * K + k]);
        f |= (fabsf(v) > SIGMA);
    }
    if (f) atomicOr(&mask[k], 1);
}

__global__ void k_compact(const int* __restrict__ mask, int* __restrict__ list,
                          int* __restrict__ count, int K) {
    if (blockIdx.x == 0 && threadIdx.x == 0) {
        int c = 0;
        for (int k = 0; k < K; ++k)
            if (mask[k]) list[c++] = k;
        *count = c;
    }
}

__device__ __forceinline__ signed char quant1(float v, float scale) {
    v = v / scale;
    v = fminf(fmaxf(v, -128.f), 127.f);
    return (signed char)(int)rintf(v);
}


__global__ void k_quant_x(const __half* __restrict__ x, signed char* __restrict__ qx,
                          float* __restrict__ srow, int K) {
    __shared__ float red[256];
    const int m = blockIdx.x;
    const __half* xr = x + (size_t)m * K;
    float mx = 0.f;
    for (int k = threadIdx.x; k < K; k += 256)
        mx = fmaxf(mx, fabsf(__half2float(xr[k])));
    red[threadIdx.x] = mx;
    __syncthreads();
    for (int s = 128; s > 0; s >>= 1) {
        if (threadIdx.x < s) red[threadIdx.x] = fmaxf(red[threadIdx.x], red[threadIdx.x + s]);
        __syncthreads();
    }
    const float scale = fmaxf(red[0] / 127.f, 1e-8f);
    if (threadIdx.x == 0) srow[m] = scale;

    const int ktiles = K >> 6;
    const size_t rowtile_base = (size_t)(m >> 4) * ktiles * 1024;
    for (int c = threadIdx.x; c < (K >> 3); c += 256) {
        const int kb = c << 3;
        union { signed char b[8]; unsigned long long u; } pk;
        #pragma unroll
        for (int j = 0; j < 8; ++j)
            pk.b[j] = quant1(__half2float(xr[kb + j]), scale);
        const int q  = (kb & 63) >> 3;
        const int hi = (q >> 1) & 1, p = ((q >> 2) << 1) | (q & 1);
        const int lane = (m & 15) + 16 * hi;
        const size_t off = rowtile_base + (size_t)(kb >> 6) * 1024 + lane * 32 + p * 8;
        *(unsigned long long*)(qx + off) = pk.u;
    }
}

__global__ void k_quant_w(const __half* __restrict__ w, const int* __restrict__ mask,
                          signed char* __restrict__ qw, float* __restrict__ scol, int K) {
    __shared__ float red[256];
    const int n = blockIdx.x;
    const __half* wr = w + (size_t)n * K;
    float mx = 0.f;
    for (int k = threadIdx.x; k < K; k += 256) {
        float v = mask[k] ? 0.f : __half2float(wr[k]);
        mx = fmaxf(mx, fabsf(v));
    }
    red[threadIdx.x] = mx;
    __syncthreads();
    for (int s = 128; s > 0; s >>= 1) {
        if (threadIdx.x < s) red[threadIdx.x] = fmaxf(red[threadIdx.x], red[threadIdx.x + s]);
        __syncthreads();
    }
    float scale = __half2float(__float2half_rn(red[0] / 127.f));
    scale = fmaxf(scale, 1e-8f);
    if (threadIdx.x == 0) scol[n] = scale;

    const int ktiles = K >> 6;
    const size_t rowtile_base = (size_t)(n >> 4) * ktiles * 1024;
    for (int c = threadIdx.x; c < (K >> 4); c += 256) {
        const int kb = c << 4;
        union { signed char b[16]; v4i v; } pk;
        #pragma unroll
        for (int j = 0; j < 16; ++j) {
            float v = mask[kb + j] ? 0.f : __half2float(wr[kb + j]);
            pk.b[j] = quant1(v, scale);
        }
        const int r  = (kb & 63) >> 4;
        const int hi = r & 1, p = r >> 1;
        const int lane = (n & 15) + 16 * hi;
        const size_t off = rowtile_base + (size_t)(kb >> 6) * 1024 + lane * 32 + p * 16;
        *(v4i*)(qw + off) = pk.v;
    }
}

__global__ __launch_bounds__(256)
void k_gemm(const signed char* __restrict__ qx, const signed char* __restrict__ qw,
            const float* __restrict__ srow, const float* __restrict__ scol,
            const __half* __restrict__ x, const __half* __restrict__ w,
            const __half* __restrict__ bias,
            const int* __restrict__ list, const int* __restrict__ count,
            __half* __restrict__ out, int M, int N, int K) {
    __shared__ float xsh[128][MAXOUT + 1];
    __shared__ float wsh[128][MAXOUT + 1];

    const int tid    = threadIdx.x;
    const int lane   = tid & 31;
    const int wave   = tid >> 5;
    const int lane15 = lane & 15;
    const int hi     = lane >> 4;
    const int wm     = wave & 1;
    const int wn     = wave >> 1;
    const int gm0    = blockIdx.y * 128;
    const int gn0    = blockIdx.x * 128;
    const int mbase  = gm0 + wm * 64;
    const int nbase  = gn0 + wn * 32;
    const int ktiles = K >> 6;

    int nout = *count; if (nout > MAXOUT) nout = MAXOUT;
    for (int idx = tid; idx < 128 * nout; idx += 256) {
        int ml = idx / nout, i = idx - ml * nout;
        xsh[ml][i] = __half2float(x[(size_t)(gm0 + ml) * K + list[i]]);
    }
    for (int idx = tid; idx < 128 * nout; idx += 256) {
        int nl = idx / nout, i = idx - nl * nout;
        wsh[nl][i] = __half2float(w[(size_t)(gn0 + nl) * K + list[i]]);
    }
    __syncthreads();

    const signed char* ap[4];
    #pragma unroll
    for (int mi = 0; mi < 4; ++mi)
        ap[mi] = qx + (size_t)((mbase >> 4) + mi) * ktiles * 1024 + lane * 32;
    const signed char* bp[2];
    #pragma unroll
    for (int ni = 0; ni < 2; ++ni)
        bp[ni] = qw + (size_t)((nbase >> 4) + ni) * ktiles * 1024 + lane * 32;

    v8i acc[4][2] = {};

    for (int kt = 0; kt < ktiles; ++kt) {
        const size_t ko = (size_t)kt * 1024;
        v8i a[4], b[2];
        #pragma unroll
        for (int mi = 0; mi < 4; ++mi) {
            v4i a0 = *(const v4i*)(ap[mi] + ko);
            v4i a1 = *(const v4i*)(ap[mi] + ko + 16);
            a[mi][0] = a0[0]; a[mi][1] = a0[1]; a[mi][2] = a0[2]; a[mi][3] = a0[3];
            a[mi][4] = a1[0]; a[mi][5] = a1[1]; a[mi][6] = a1[2]; a[mi][7] = a1[3];
            __builtin_prefetch(ap[mi] + ko + 1024, 0, 0);
        }
        #pragma unroll
        for (int ni = 0; ni < 2; ++ni) {
            v4i b0 = *(const v4i*)(bp[ni] + ko);
            v4i b1 = *(const v4i*)(bp[ni] + ko + 16);
            b[ni][0] = b0[0]; b[ni][1] = b0[1]; b[ni][2] = b0[2]; b[ni][3] = b0[3];
            b[ni][4] = b1[0]; b[ni][5] = b1[1]; b[ni][6] = b1[2]; b[ni][7] = b1[3];
            __builtin_prefetch(bp[ni] + ko + 1024, 0, 0);
        }
        #pragma unroll
        for (int mi = 0; mi < 4; ++mi)
            #pragma unroll
            for (int ni = 0; ni < 2; ++ni)
                acc[mi][ni] = __builtin_amdgcn_wmma_i32_16x16x64_iu8(
                     true, a[mi],  true, b[ni],
                    acc[mi][ni],  false,  false);
    }

    #pragma unroll
    for (int mi = 0; mi < 4; ++mi) {
        #pragma unroll
        for (int ni = 0; ni < 2; ++ni) {
            const int n  = nbase + ni * 16 + lane15;
            const int nl = n - gn0;
            const float sc = scol[n];
            const float bv = __half2float(bias[n]);
            #pragma unroll
            for (int j = 0; j < 8; ++j) {
                const int m  = mbase + mi * 16 + j + 8 * hi;
                const int ml = m - gm0;
                float f = (float)acc[mi][ni][j] * srow[m] * sc;
                for (int i = 0; i < nout; ++i)
                    f += xsh[ml][i] * wsh[nl][i];
                f += bv;
                out[(size_t)m * N + n] = __float2half(f);
            }
        }
    }
}

extern "C" void kernel_launch(void* const* d_in, const int* in_sizes, int n_in,
                              void* d_out, int out_size, void* d_ws, size_t ws_size,
                              hipStream_t stream) {
    const __half* x    = (const __half*)d_in[0];
    const __half* w    = (const __half*)d_in[1];
    const __half* bias = (const __half*)d_in[2];
    __half* out = (__half*)d_out;

    const int N = in_sizes[2];
    const int K = in_sizes[1] / N;
    const int M = in_sizes[0] / K;

    char* ws = (char*)d_ws;
    signed char* qx = (signed char*)ws;  ws += (size_t)M * K;
    signed char* qw = (signed char*)ws;  ws += (size_t)N * K;
    float* srow = (float*)ws;            ws += (size_t)M * sizeof(float);
    float* scol = (float*)ws;            ws += (size_t)N * sizeof(float);
    int* mask   = (int*)ws;              ws += (size_t)K * sizeof(int);
    int* list   = (int*)ws;              ws += (size_t)K * sizeof(int);
    int* count  = (int*)ws;

    k_init<<<(K + 255) / 256, 256, 0, stream>>>(mask, K);

    const int ROWS = 512;
    dim3 og((K + 255) / 256, (M + ROWS - 1) / ROWS);
    k_outlier<<<og, 256, 0, stream>>>(x, mask, M, K, ROWS);

    k_compact<<<1, 1, 0, stream>>>(mask, list, count, K);

    k_quant_x<<<M, 256, 0, stream>>>(x, qx, srow, K);
    k_quant_w<<<N, 256, 0, stream>>>(w, mask, qw, scol, K);

    dim3 gg(N / 128, M / 128);
    k_gemm<<<gg, 256, 0, stream>>>(qx, qw, srow, scol, x, w, bias,
                                   list, count, out, M, N, K);
}
